// NVFP4FakeQuantizedGptOssExperts_38912403702499
// MI455X (gfx1250) — compile-verified, matrix operand order checked by reading, not run on hardware
//
#include <hip/hip_runtime.h>
#include <stdint.h>

typedef __attribute__((ext_vector_type(16))) _Float16 v16h;
typedef __attribute__((ext_vector_type(8)))  float    v8f;
typedef unsigned short u16;
typedef int i32x4_vec __attribute__((vector_size(16)));

union F16Frag { v16h v; uint4 q[2]; };

#define T_TOK 1024
#define KDIM  1024
#define NEXP  8

// ---- CDNA5 async global->LDS copy (no VGPR staging, tracked by ASYNCcnt) ----
// param0: int4* in AS(1) (prints as "__device__"); param1: LDS-side pointer.
#define AS1I4(p) ((__attribute__((address_space(1))) i32x4_vec*)(uintptr_t)(p))
#define AS3I4(p) ((__attribute__((address_space(3))) i32x4_vec*)(unsigned int)(uintptr_t)(p))

#if __has_builtin(__builtin_amdgcn_global_load_async_to_lds_b128)
#define USE_ASYNC 1
#if __has_builtin(__builtin_amdgcn_s_wait_asynccnt)
#define WAIT_ASYNC() __builtin_amdgcn_s_wait_asynccnt(0)
#else
#define WAIT_ASYNC() asm volatile("s_wait_asynccnt 0" ::: "memory")
#endif
#else
#define USE_ASYNC 0
#endif

// ---------------- quantization helpers (exact RNE low-precision rounding) ---
__device__ __forceinline__ float floor_log2_pos(float ax) {
  return (float)((int)((__float_as_uint(ax) >> 23) & 0xFF) - 127);
}
__device__ __forceinline__ float qrne(float x, float elo, float ehi, float mb, float mx) {
  float ax = fabsf(x);
  float e  = floor_log2_pos(fmaxf(ax, 1e-30f));
  e = fminf(fmaxf(e, elo), ehi);
  float step = exp2f(e - mb);                    // exact (integer exponent)
  float q = fminf(rintf(ax / step) * step, mx);  // rintf = round half to even
  return copysignf(q, x);
}
__device__ __forceinline__ float q_e4m3(float x) { return qrne(x, -6.f, 8.f, 3.f, 448.f); }
__device__ __forceinline__ float q_e2m1(float x) { return qrne(x,  0.f, 2.f, 1.f,   6.f); }
__device__ __forceinline__ float pt_of(float amax) { return amax > 0.f ? amax * (1.f / 2688.f) : 1.f; }

__device__ __forceinline__ void atomicMaxPos(float* addr, float v) {
  atomicMax((unsigned int*)addr, __float_as_uint(v));  // non-negative floats only
}
__device__ __forceinline__ float block_max(float v, float* red) {
  int tid = threadIdx.x;
  red[tid] = v; __syncthreads();
  for (int s = 128; s > 0; s >>= 1) {
    if (tid < s) red[tid] = fmaxf(red[tid], red[tid + s]);
    __syncthreads();
  }
  return red[0];
}

// ---------------- init: mask / combine weights / zero amax ------------------
__global__ __launch_bounds__(256) void init_kernel(const int* __restrict__ ri,
                                                   const float* __restrict__ rw,
                                                   float* __restrict__ maskf,
                                                   float* __restrict__ wgt,
                                                   float* __restrict__ amax) {
  int id = blockIdx.x * 256 + threadIdx.x;
  if (id < 32) amax[id] = 0.f;
  if (id >= NEXP * T_TOK) return;
  int e = id >> 10, t = id & 1023;
  int r0 = ri[2 * t], r1 = ri[2 * t + 1];
  float m = (r0 == e || r1 == e) ? 1.f : 0.f;
  maskf[id] = m;
  wgt[id] = rw[t * NEXP + e] * m;
}

// ---------------- amax reductions ------------------------------------------
__global__ __launch_bounds__(256) void amax_w_kernel(const float* __restrict__ W,
                                                     float* __restrict__ amax, int nPerE) {
  __shared__ float red[256];
  int e = blockIdx.y;
  const float* p = W + (size_t)e * nPerE + (size_t)(blockIdx.x * 256 + threadIdx.x) * 16;
  float m = 0.f;
#pragma unroll
  for (int c = 0; c < 4; ++c) {
    float4 v = ((const float4*)p)[c];
    m = fmaxf(m, fmaxf(fmaxf(fabsf(v.x), fabsf(v.y)), fmaxf(fabsf(v.z), fabsf(v.w))));
  }
  float bm = block_max(m, red);
  if (threadIdx.x == 0) atomicMaxPos(&amax[e], bm);
}

__global__ __launch_bounds__(256) void amax_h_kernel(const float* __restrict__ h,
                                                     const float* __restrict__ maskf,
                                                     float* __restrict__ amax) {
  __shared__ float red[256];
  int e = blockIdx.y;
  size_t base = (size_t)(blockIdx.x * 256 + threadIdx.x) * 16;
  float m = 0.f;
#pragma unroll
  for (int c = 0; c < 4; ++c) {
    size_t idx = base + 4 * c;
    float4 v = *(const float4*)(h + idx);
    int t = (int)(idx >> 10);
    float mk = maskf[e * T_TOK + t];
    float a = fmaxf(fmaxf(fabsf(v.x), fabsf(v.y)), fmaxf(fabsf(v.z), fabsf(v.w)));
    m = fmaxf(m, a * mk);
  }
  float bm = block_max(m, red);
  if (threadIdx.x == 0) atomicMaxPos(&amax[e], bm);
}

// ---------------- weight quantize: fp32 [E][K][Nc] -> f16 (s_blk*f4) [E][Nc][K]
__global__ __launch_bounds__(256) void quant_w_kernel(const float* __restrict__ W,
                                                      _Float16* __restrict__ Wt,
                                                      const float* __restrict__ amaxW,
                                                      int K, int Nc) {
  int e = blockIdx.y;
  int id = blockIdx.x * 256 + threadIdx.x;
  int nb16 = Nc / 16;
  if (id >= K * nb16) return;
  int k = id / nb16, nb = id % nb16;
  const float* p = W + ((size_t)e * K + k) * Nc + nb * 16;
  float x[16];
#pragma unroll
  for (int c = 0; c < 4; ++c) {
    float4 v = ((const float4*)p)[c];
    x[4 * c] = v.x; x[4 * c + 1] = v.y; x[4 * c + 2] = v.z; x[4 * c + 3] = v.w;
  }
  float pt = pt_of(amaxW[e]);
  float bm = 0.f;
#pragma unroll
  for (int i = 0; i < 16; ++i) bm = fmaxf(bm, fabsf(x[i]));
  float sbs = q_e4m3(fminf(fmaxf(bm * (1.f / 6.f) / pt, 0.015625f), 448.f));
  float inv = 1.f / (pt * sbs);
  _Float16* o = Wt + ((size_t)e * Nc + (size_t)nb * 16) * K + k;  // transposed scatter
#pragma unroll
  for (int i = 0; i < 16; ++i) {
    float f4 = q_e2m1(fminf(fmaxf(x[i] * inv, -6.f), 6.f));
    o[(size_t)i * K] = (_Float16)(f4 * sbs);
  }
}

// ---------------- activation quantize: fp32 rows -> f16 rows (contiguous) ---
__global__ __launch_bounds__(256) void quant_a_kernel(const float* __restrict__ X, int perExpert,
                                                      _Float16* __restrict__ Xq,
                                                      const float* __restrict__ amaxX, int Kc) {
  int e = blockIdx.y;
  int id = blockIdx.x * 256 + threadIdx.x;
  int nb16 = Kc / 16;
  if (id >= T_TOK * nb16) return;
  int row = id / nb16, bb = id % nb16;
  const float* p = X + (perExpert ? (size_t)e * T_TOK * Kc : 0) + (size_t)row * Kc + bb * 16;
  float x[16];
#pragma unroll
  for (int c = 0; c < 4; ++c) {
    float4 v = ((const float4*)p)[c];
    x[4 * c] = v.x; x[4 * c + 1] = v.y; x[4 * c + 2] = v.z; x[4 * c + 3] = v.w;
  }
  float pt = pt_of(amaxX[e]);
  float bm = 0.f;
#pragma unroll
  for (int i = 0; i < 16; ++i) bm = fmaxf(bm, fabsf(x[i]));
  float sbs = q_e4m3(fminf(fmaxf(bm * (1.f / 6.f) / pt, 0.015625f), 448.f));
  float inv = 1.f / (pt * sbs);
  union { _Float16 hf[16]; uint4 q[2]; } pk;
#pragma unroll
  for (int i = 0; i < 16; ++i) {
    float f4 = q_e2m1(fminf(fmaxf(x[i] * inv, -6.f), 6.f));
    pk.hf[i] = (_Float16)(f4 * sbs);
  }
  uint4* o = (uint4*)(Xq + ((size_t)e * T_TOK + row) * Kc + bb * 16);
  o[0] = pk.q[0]; o[1] = pk.q[1];
}

// ---------------- WMMA GEMM: C = (A @ B^T_stored) * ptA*ptB + bias, fused epilogue
// A: f16 [E][T][1024] row-major; B: f16 [E][N][1024] (pre-transposed weights)
// EPI==0: clipped-GLU -> inter [E][T][N/2] + masked amax_i
// EPI==1: +bias, * routing weight -> outs [E][T][N]
template <int EPI>
__global__ __launch_bounds__(256) void gemm_kernel(const _Float16* __restrict__ Aq,
                                                   const _Float16* __restrict__ Bt,
                                                   const float* __restrict__ bias,
                                                   float* __restrict__ Out,
                                                   const float* __restrict__ amaxA,
                                                   const float* __restrict__ amaxB,
                                                   const float* __restrict__ maskf,
                                                   const float* __restrict__ wgt,
                                                   float* __restrict__ amaxOut, int N) {
  constexpr int BM = 128, BN = 128, BK = 64;
  __shared__ __align__(16) char smem[4 * 128 * 72 * 2];  // A[2]|B[2], 72-half padded rows
  __shared__ float red[256];
  const int e = blockIdx.z;
  const int m0 = blockIdx.y * BM, n0 = blockIdx.x * BN;
  const int tid = threadIdx.x, lane = tid & 31, wave = tid >> 5;
  const int wm = wave >> 2, wn = wave & 3;       // 2 x 4 wave grid, wave tile 64x32
  const int l = lane & 15, hi = lane >> 4;

  const _Float16* A = Aq + (size_t)e * T_TOK * KDIM;
  const _Float16* B = Bt + (size_t)e * (size_t)N * KDIM;
  u16* sA = (u16*)smem;                 // [2][128][72]
  u16* sB = (u16*)(smem + 2 * 128 * 72 * 2);

  v8f acc[4][2];
#pragma unroll
  for (int i = 0; i < 4; ++i)
#pragma unroll
    for (int j = 0; j < 2; ++j)
#pragma unroll
      for (int r = 0; r < 8; ++r) acc[i][j][r] = 0.f;

  const int KT = KDIM / BK;  // 16

#if USE_ASYNC
  // ---- zero-VGPR staging via GLOBAL_LOAD_ASYNC_TO_LDS_B128 (ASYNCcnt) ----
  auto issue = [&](int kt, int nbuf) {
#pragma unroll
    for (int i = 0; i < 4; ++i) {
      int c = tid + i * 256, row = c >> 3, c8 = c & 7;
      const _Float16* pa = A + (size_t)(m0 + row) * KDIM + kt * BK + c8 * 8;
      const _Float16* pb = B + (size_t)(n0 + row) * KDIM + kt * BK + c8 * 8;
      u16* la = sA + nbuf * 9216 + row * 72 + c8 * 8;
      u16* lb = sB + nbuf * 9216 + row * 72 + c8 * 8;
      __builtin_amdgcn_global_load_async_to_lds_b128(AS1I4(pa), AS3I4(la), 0, 0);
      __builtin_amdgcn_global_load_async_to_lds_b128(AS1I4(pb), AS3I4(lb), 0, 0);
    }
  };
  issue(0, 0);
  WAIT_ASYNC();
  __syncthreads();
#else
  // ---- fallback: VGPR-staged double buffering ----
  uint4 ra[4], rb[4];
  auto fetch = [&](int kt) {
#pragma unroll
    for (int i = 0; i < 4; ++i) {
      int c = tid + i * 256, row = c >> 3, c8 = c & 7;
      ra[i] = *(const uint4*)(A + (size_t)(m0 + row) * KDIM + kt * BK + c8 * 8);
      rb[i] = *(const uint4*)(B + (size_t)(n0 + row) * KDIM + kt * BK + c8 * 8);
    }
  };
  auto stash = [&](int buf) {
#pragma unroll
    for (int i = 0; i < 4; ++i) {
      int c = tid + i * 256, row = c >> 3, c8 = c & 7;
      *(uint4*)(sA + buf * 9216 + row * 72 + c8 * 8) = ra[i];
      *(uint4*)(sB + buf * 9216 + row * 72 + c8 * 8) = rb[i];
    }
  };
  fetch(0); stash(0); __syncthreads();
#endif

  for (int kt = 0; kt < KT; ++kt) {
    int buf = kt & 1;
#if USE_ASYNC
    if (kt + 1 < KT) issue(kt + 1, buf ^ 1);   // safe: buf^1 readers done at prior barrier
#else
    if (kt + 1 < KT) fetch(kt + 1);
#endif
#pragma unroll
    for (int ks = 0; ks < 2; ++ks) {
      F16Frag afr[4], bfr[2];
#pragma unroll
      for (int fm = 0; fm < 4; ++fm) {
        const u16* p = sA + buf * 9216 + (wm * 64 + fm * 16 + l) * 72 + ks * 32 + hi * 8;
        afr[fm].q[0] = *(const uint4*)p;         // K 0-7   (8-15 for hi lanes)
        afr[fm].q[1] = *(const uint4*)(p + 16);  // K 16-23 (24-31)
      }
#pragma unroll
      for (int fn = 0; fn < 2; ++fn) {
        const u16* p = sB + buf * 9216 + (wn * 32 + fn * 16 + l) * 72 + ks * 32 + hi * 8;
        bfr[fn].q[0] = *(const uint4*)p;
        bfr[fn].q[1] = *(const uint4*)(p + 16);
      }
#pragma unroll
      for (int fm = 0; fm < 4; ++fm)
#pragma unroll
        for (int fn = 0; fn < 2; ++fn)
          acc[fm][fn] = __builtin_amdgcn_wmma_f32_16x16x32_f16(
              false, afr[fm].v, false, bfr[fn].v, (short)0, acc[fm][fn], false, false);
    }
#if USE_ASYNC
    if (kt + 1 < KT) { WAIT_ASYNC(); __syncthreads(); }
#else
    if (kt + 1 < KT) { __syncthreads(); stash((kt + 1) & 1); __syncthreads(); }
#endif
  }

  float s = pt_of(amaxA[e]) * pt_of(amaxB[e]);   // fold per-tensor scales
  const float* bi = bias + (size_t)e * N;

  if (EPI == 1) {
    float* O = Out + (size_t)e * T_TOK * N;
#pragma unroll
    for (int fm = 0; fm < 4; ++fm) {
      int rbase = m0 + wm * 64 + fm * 16 + hi * 8;
#pragma unroll
      for (int fn = 0; fn < 2; ++fn) {
        int col = n0 + wn * 32 + fn * 16 + l;
        float bv = bi[col];
#pragma unroll
        for (int r = 0; r < 8; ++r) {
          int row = rbase + r;
          O[(size_t)row * N + col] = (acc[fm][fn][r] * s + bv) * wgt[e * T_TOK + row];
        }
      }
    }
  } else {
    __syncthreads();                       // staging fully consumed; reuse as C bounce
    float* lC = (float*)smem;              // [128][130]
#pragma unroll
    for (int fm = 0; fm < 4; ++fm) {
      int rr0 = wm * 64 + fm * 16 + hi * 8;
#pragma unroll
      for (int fn = 0; fn < 2; ++fn) {
        int cc = wn * 32 + fn * 16 + l;
        float bv = bi[n0 + cc];
#pragma unroll
        for (int r = 0; r < 8; ++r) lC[(rr0 + r) * 130 + cc] = acc[fm][fn][r] * s + bv;
      }
    }
    __syncthreads();
    float lmax = 0.f;
    float* O = Out + (size_t)e * T_TOK * (N / 2);
#pragma unroll 4
    for (int i = 0; i < 32; ++i) {
      int idx = tid + i * 256;
      int row = idx >> 6, j = idx & 63;                  // 128 x 64 inter tile
      float g = lC[row * 130 + 2 * j];
      float u = lC[row * 130 + 2 * j + 1];
      g = fminf(g, 7.f);
      u = fminf(fmaxf(u, -7.f), 7.f);
      float glu = g / (1.f + __expf(-1.702f * g));       // g * sigmoid(1.702 g)
      float v = (u + 1.f) * glu;
      O[(size_t)(m0 + row) * (N / 2) + (n0 >> 1) + j] = v;
      lmax = fmaxf(lmax, fabsf(v) * maskf[e * T_TOK + m0 + row]);
    }
    float bm = block_max(lmax, red);
    if (tid == 0) atomicMaxPos(&amaxOut[e], bm);
  }
}

// ---------------- combine: out[t,h] = sum_e outs_weighted[e][t][h] ----------
__global__ __launch_bounds__(256) void combine_kernel(const float* __restrict__ outsW,
                                                      float* __restrict__ out) {
  int id = blockIdx.x * 256 + threadIdx.x;   // over T*H/4
  float4 s = make_float4(0.f, 0.f, 0.f, 0.f);
#pragma unroll
  for (int e = 0; e < NEXP; ++e) {
    float4 v = ((const float4*)(outsW + (size_t)e * (T_TOK * KDIM)))[id];
    s.x += v.x; s.y += v.y; s.z += v.z; s.w += v.w;
  }
  ((float4*)out)[id] = s;
}

extern "C" void kernel_launch(void* const* d_in, const int* in_sizes, int n_in,
                              void* d_out, int out_size, void* d_ws, size_t ws_size,
                              hipStream_t stream) {
  const float* h  = (const float*)d_in[0];   // [2,512,1024] -> [1024][1024]
  const int*   ri = (const int*)d_in[1];     // [1024][2]
  const float* rw = (const float*)d_in[2];   // [1024][8]
  const float* w1 = (const float*)d_in[3];   // [8][1024][2048]
  const float* b1 = (const float*)d_in[4];   // [8][2048]
  const float* w2 = (const float*)d_in[5];   // [8][1024][1024]
  const float* b2 = (const float*)d_in[6];   // [8][1024]
  float* out = (float*)d_out;

  char* ws = (char*)d_ws;
  // scratch layout (buffers aliased across dead ranges; total ~96MB + 65KB)
  _Float16* W1V = (_Float16*)(ws);                         // 32MB  [8][2048][1024] f16
  _Float16* W2V = (_Float16*)(ws + (32ull << 20));         // 16MB  [8][1024][1024] f16
  _Float16* FQA = (_Float16*)(ws + (48ull << 20));         // 16MB  fqh, then fqi
  float*    INT = (float*)(ws + (64ull << 20));            // 32MB  inter, then weighted outs
  float*    AMX = (float*)(ws + (96ull << 20));            // [0:8)=h [8:16)=w1 [16:24)=w2 [24:32)=i
  float*    MSK = (float*)(ws + (96ull << 20) + 1024);     // [8][1024]
  float*    WGT = (float*)(ws + (96ull << 20) + 1024 + 32768);

  init_kernel<<<32, 256, 0, stream>>>(ri, rw, MSK, WGT, AMX);

  amax_w_kernel<<<dim3(512, 8), 256, 0, stream>>>(w1, AMX + 8, 1024 * 2048);
  amax_w_kernel<<<dim3(256, 8), 256, 0, stream>>>(w2, AMX + 16, 1024 * 1024);
  amax_h_kernel<<<dim3(256, 8), 256, 0, stream>>>(h, MSK, AMX + 0);

  quant_w_kernel<<<dim3(512, 8), 256, 0, stream>>>(w1, W1V, AMX + 8, 1024, 2048);
  quant_w_kernel<<<dim3(256, 8), 256, 0, stream>>>(w2, W2V, AMX + 16, 1024, 1024);
  quant_a_kernel<<<dim3(256, 8), 256, 0, stream>>>(h, 0, FQA, AMX + 0, 1024);

  gemm_kernel<0><<<dim3(16, 8, 8), 256, 0, stream>>>(FQA, W1V, b1, INT,
                                                     AMX + 0, AMX + 8, MSK, WGT, AMX + 24, 2048);

  quant_a_kernel<<<dim3(256, 8), 256, 0, stream>>>(INT, 1, FQA, AMX + 24, 1024);

  gemm_kernel<1><<<dim3(8, 8, 8), 256, 0, stream>>>(FQA, W2V, b2, INT,
                                                    AMX + 24, AMX + 16, MSK, WGT, nullptr, 1024);

  combine_kernel<<<1024, 256, 0, stream>>>(INT, out);
}
